// CrossAttention_83004537962801
// MI455X (gfx1250) — matrix operand order repaired and checked by reading, not compiled, not run on hardware
//
#include <hip/hip_runtime.h>
#include <hip/hip_bf16.h>
#include <cstdint>

// ---------------------------------------------------------------------------
// CrossAttention block for MI455X (gfx1250, wave32, WMMA).
// GEMMs: v_wmma_f32_16x16x32_bf16, LDS tiles streamed with
// global_load_async_to_lds_b128 (ASYNCcnt) and double buffering.
// ---------------------------------------------------------------------------

typedef __bf16 bf16;
typedef __attribute__((ext_vector_type(16))) __bf16 v16bf;
typedef __attribute__((ext_vector_type(8)))  float  v8f;
typedef __attribute__((ext_vector_type(4)))  int    v4i;

#define B_   16
#define C_   512
#define HW_  4096
#define S_   256
#define CTX_ 768
#define GRP_ 32
#define EPS_ 1e-5f

#define AS1 __attribute__((address_space(1)))
#define AS3 __attribute__((address_space(3)))

// ---- gfx1250 async memory->LDS copy (16 bytes), with safe fallback --------
// Toolchain probe: builtin parameters are pointers to 16-byte int vectors.
__device__ __forceinline__ void async_b128(const void* g, void* lds) {
#if __has_builtin(__builtin_amdgcn_global_load_async_to_lds_b128)
  __builtin_amdgcn_global_load_async_to_lds_b128(
      (AS1 v4i*)(uintptr_t)g, (AS3 v4i*)(uint32_t)(uintptr_t)lds, 0, 0);
#else
  *(float4*)lds = *(const float4*)g;
#endif
}

__device__ __forceinline__ void wait_async() {
#if __has_builtin(__builtin_amdgcn_s_wait_asynccnt)
  __builtin_amdgcn_s_wait_asynccnt(0);
#elif defined(__gfx1250__)
  asm volatile("s_wait_asynccnt 0x0" ::: "memory");
#endif
}

// ---- WMMA fragment loaders (ISA 7.12.2 layouts, wave32) -------------------
// A-matrix 16x32 bf16: lane row = lane&15, K base = (lane>>4)*8,
// elements 0..7 -> K=base..base+7, elements 8..15 -> K=base+16..base+23.
__device__ __forceinline__ v16bf frag_a(const bf16* base, int lda) {
  int lane = threadIdx.x & 31;
  const bf16* p = base + (lane & 15) * lda + ((lane >> 4) << 3);
  union { v16bf v; float4 q[2]; } u;
  u.q[0] = *(const float4*)(p);
  u.q[1] = *(const float4*)(p + 16);
  return u.v;
}
// B-matrix 32x16 bf16 from W[N][K] row-major: lane col = lane&15,
// K base = (lane>>4)*8, elements 0..7 -> K=base..base+7, 8..15 -> K=base+16..base+23.
__device__ __forceinline__ v16bf frag_b(const bf16* base, int ldb) {
  int lane = threadIdx.x & 31;
  const bf16* p = base + (lane & 15) * ldb + ((lane >> 4) << 3);
  union { v16bf v; float4 q[2]; } u;
  u.q[0] = *(const float4*)(p);
  u.q[1] = *(const float4*)(p + 16);
  return u.v;
}

__device__ __forceinline__ v8f wmma_bf16(v16bf a, v16bf b, v8f c) {
  return __builtin_amdgcn_wmma_f32_16x16x32_bf16(false, a, false, b, (short)0, c,
                                                 false, false);
}

// ---- small utility kernels ------------------------------------------------

__global__ void cvt_bf16_kernel(const float* __restrict__ in,
                                bf16* __restrict__ out, int n) {
  int i = blockIdx.x * blockDim.x + threadIdx.x;
  if (i < n) out[i] = (bf16)in[i];
}

// One block per (b, group): reduce 16 channels x 4096 pixels = 65536 floats.
__global__ __launch_bounds__(256) void gn_stats_kernel(
    const float* __restrict__ x, float* __restrict__ mean,
    float* __restrict__ rstd) {
  int bg = blockIdx.x;  // 0..511, b-major (x layout [b][c][hw], group contig)
  const float4* p4 = (const float4*)(x + (size_t)bg * 65536);
  float s = 0.f, ss = 0.f;
  for (int i = threadIdx.x; i < 16384; i += 256) {
    float4 v = p4[i];
    s  += v.x + v.y + v.z + v.w;
    ss += v.x * v.x + v.y * v.y + v.z * v.z + v.w * v.w;
  }
  for (int o = 16; o > 0; o >>= 1) {
    s  += __shfl_down(s, o, 32);
    ss += __shfl_down(ss, o, 32);
  }
  __shared__ float buf[2][8];
  int w = threadIdx.x >> 5, lane = threadIdx.x & 31;
  if (lane == 0) { buf[0][w] = s; buf[1][w] = ss; }
  __syncthreads();
  if (threadIdx.x == 0) {
    float S = 0.f, SS = 0.f;
    for (int i = 0; i < 8; i++) { S += buf[0][i]; SS += buf[1][i]; }
    float mu  = S * (1.f / 65536.f);
    float var = SS * (1.f / 65536.f) - mu * mu;
    mean[bg] = mu;
    rstd[bg] = rsqrtf(var + EPS_);
  }
}

// Normalize x and transpose [b][c][hw] f32 -> [b][hw][c] bf16 (LDS tiled).
__global__ __launch_bounds__(256) void gn_pack_kernel(
    const float* __restrict__ x, const float* __restrict__ gw,
    const float* __restrict__ gb, const float* __restrict__ mean,
    const float* __restrict__ rstd, bf16* __restrict__ xnT) {
  __shared__ bf16 tile[64][72];  // padded vs bank conflicts
  int b = blockIdx.z, c0 = blockIdx.y * 64, p0 = blockIdx.x * 64;
  const float* xb = x + (size_t)b * C_ * HW_;
#pragma unroll 4
  for (int j = 0; j < 16; j++) {
    int linear = j * 256 + threadIdx.x;
    int cl = linear >> 6, pl = linear & 63;
    int c = c0 + cl;
    int g = c >> 4;
    float mu = mean[b * GRP_ + g], rs = rstd[b * GRP_ + g];
    float v = xb[(size_t)c * HW_ + p0 + pl];
    tile[pl][cl] = (bf16)((v - mu) * rs * gw[c] + gb[c]);
  }
  __syncthreads();
  bf16* o = xnT + ((size_t)b * HW_ + p0) * C_ + c0;
#pragma unroll 4
  for (int j = 0; j < 16; j++) {
    int linear = j * 256 + threadIdx.x;
    int pl = linear >> 6, cl = linear & 63;
    o[(size_t)pl * C_ + cl] = tile[pl][cl];
  }
}

// LayerNorm a 768-wide context row and pack to bf16. One block per row.
__global__ __launch_bounds__(256) void ln_pack_kernel(
    const float* __restrict__ ctx, const float* __restrict__ w,
    const float* __restrict__ bcoef, bf16* __restrict__ out) {
  int row = blockIdx.x;
  const float* p = ctx + (size_t)row * CTX_;
  float s = 0.f, ss = 0.f;
  for (int i = threadIdx.x; i < CTX_; i += 256) {
    float v = p[i]; s += v; ss += v * v;
  }
  for (int o = 16; o > 0; o >>= 1) {
    s += __shfl_down(s, o, 32); ss += __shfl_down(ss, o, 32);
  }
  __shared__ float buf[2][8];
  __shared__ float stats[2];
  int wv = threadIdx.x >> 5, lane = threadIdx.x & 31;
  if (lane == 0) { buf[0][wv] = s; buf[1][wv] = ss; }
  __syncthreads();
  if (threadIdx.x == 0) {
    float S = 0.f, SS = 0.f;
    for (int i = 0; i < 8; i++) { S += buf[0][i]; SS += buf[1][i]; }
    float mu = S * (1.f / CTX_);
    stats[0] = mu;
    stats[1] = rsqrtf(SS * (1.f / CTX_) - mu * mu + EPS_);
  }
  __syncthreads();
  float mu = stats[0], rs = stats[1];
  bf16* o = out + (size_t)row * CTX_;
  for (int i = threadIdx.x; i < CTX_; i += 256)
    o[i] = (bf16)((p[i] - mu) * rs * w[i] + bcoef[i]);
}

// ---- generic WMMA GEMM: out[b][M][N] (bf16) = A[b][M][K] * W[N][K]^T + bias
// Block tile 64x64, 8 waves; double-buffered async LDS staging.
__global__ __launch_bounds__(256) void gemm_bf16_nt_kernel(
    const bf16* __restrict__ A, long strideA, const bf16* __restrict__ W,
    const float* __restrict__ bias, bf16* __restrict__ out, int ldo,
    long strideO, int K) {
  __shared__ __align__(16) bf16 As[2][64][64];
  __shared__ __align__(16) bf16 Ws[2][64][64];
  const bf16* Ab = A + strideA * blockIdx.z;
  bf16* Ob = out + strideO * blockIdx.z;
  int row0 = blockIdx.x * 64, col0 = blockIdx.y * 64;
  int tid = threadIdx.x, lane = tid & 31, wid = tid >> 5;
  int wm = wid & 3, wn = wid >> 2;
  int r = tid >> 2, seg = tid & 3;
  const bf16* arow = Ab + (size_t)(row0 + r) * K + seg * 16;
  const bf16* wrow = W + (size_t)(col0 + r) * K + seg * 16;

  // prologue: stage chunk 0 into buffer 0
  async_b128(arow,     &As[0][r][seg * 16]);
  async_b128(arow + 8, &As[0][r][seg * 16 + 8]);
  async_b128(wrow,     &Ws[0][r][seg * 16]);
  async_b128(wrow + 8, &Ws[0][r][seg * 16 + 8]);
  v8f c0 = {}, c1 = {};
  const int nch = K >> 6;
  wait_async();
  __syncthreads();
  for (int i = 0; i < nch; ++i) {
    const int cur = i & 1;
    if (i + 1 < nch) {  // stream next chunk while computing this one
      const bf16* a2 = arow + (i + 1) * 64;
      const bf16* w2 = wrow + (i + 1) * 64;
      async_b128(a2,     &As[cur ^ 1][r][seg * 16]);
      async_b128(a2 + 8, &As[cur ^ 1][r][seg * 16 + 8]);
      async_b128(w2,     &Ws[cur ^ 1][r][seg * 16]);
      async_b128(w2 + 8, &Ws[cur ^ 1][r][seg * 16 + 8]);
    }
    if (i + 2 < nch) __builtin_prefetch(arow + (i + 2) * 64, 0, 1);
#pragma unroll
    for (int kk = 0; kk < 64; kk += 32) {
      v16bf a  = frag_a(&As[cur][wm * 16][kk], 64);
      v16bf b0 = frag_b(&Ws[cur][wn * 32][kk], 64);
      v16bf b1 = frag_b(&Ws[cur][wn * 32 + 16][kk], 64);
      c0 = wmma_bf16(a, b0, c0);
      c1 = wmma_bf16(a, b1, c1);
    }
    wait_async();
    __syncthreads();
  }
  int n = lane & 15, mo = (lane >> 4) * 8;
  int cA = col0 + wn * 32 + n, cB = cA + 16;
  float biasA = bias[cA], biasB = bias[cB];
#pragma unroll
  for (int rr = 0; rr < 8; rr++) {
    size_t row = (size_t)(row0 + wm * 16 + mo + rr);
    Ob[row * ldo + cA] = (bf16)(c0[rr] + biasA);
    Ob[row * ldo + cB] = (bf16)(c1[rr] + biasB);
  }
}

// ---- fused attention: 32 q-rows per block, full S=256, softmax in LDS -----
__global__ __launch_bounds__(256) void attn_kernel(
    const bf16* __restrict__ q,   // [b][4096][512]
    const bf16* __restrict__ kv,  // [b][256][1024] (k=0..511, v=512..1023)
    bf16* __restrict__ aout) {    // [b][4096][512]
  __shared__ __align__(16) bf16 Qs[32][64];             // 4 KB
  __shared__ __align__(16) unsigned char slab[32768];   // Ks | sim | Vt
  __shared__ __align__(16) bf16 attnbuf[32][256];       // 16 KB
  bf16  (*Ks)[64]   = (bf16(*)[64])(void*)slab;   // [256][64]
  float (*sim)[256] = (float(*)[256])(void*)slab; // [32][256]
  bf16  (*Vt)[64]   = (bf16(*)[64])(void*)slab;   // [256][64]

  int b = blockIdx.y, p0 = blockIdx.x * 32;
  const bf16* qb  = q  + ((size_t)b * HW_ + p0) * C_;
  const bf16* kvb = kv + (size_t)b * S_ * (2 * C_);
  int tid = threadIdx.x, lane = tid & 31, wid = tid >> 5;
  int wr = wid & 1, wc = wid >> 1;  // wave: rows wr*16, cols wc*64

  // ---- sim = q @ k^T ----
  v8f c[4] = {{}, {}, {}, {}};
  for (int kc = 0; kc < C_; kc += 64) {
    {
      int rq = tid >> 3, sq = tid & 7;
      async_b128(&qb[(size_t)rq * C_ + kc + sq * 8], &Qs[rq][sq * 8]);
    }
#pragma unroll
    for (int j = 0; j < 8; j++) {
      int fid = j * 256 + tid;
      int s = fid >> 3, seg = fid & 7;
      async_b128(&kvb[(size_t)s * (2 * C_) + kc + seg * 8], &Ks[s][seg * 8]);
    }
    wait_async();
    __syncthreads();
#pragma unroll
    for (int kk = 0; kk < 64; kk += 32) {
      v16bf a = frag_a(&Qs[wr * 16][kk], 64);
#pragma unroll
      for (int t = 0; t < 4; t++) {
        v16bf bb = frag_b(&Ks[wc * 64 + t * 16][kk], 64);
        c[t] = wmma_bf16(a, bb, c[t]);
      }
    }
    __syncthreads();
  }
  {  // spill scaled sim to LDS (reuses Ks slab)
    const float scale = 0.044194173824159216f;  // 512^-0.5
    int n = lane & 15, mo = (lane >> 4) * 8;
#pragma unroll
    for (int t = 0; t < 4; t++)
#pragma unroll
      for (int r = 0; r < 8; r++)
        sim[wr * 16 + mo + r][wc * 64 + t * 16 + n] = c[t][r] * scale;
  }
  __syncthreads();

  // ---- softmax: 8 lanes per row, 32 elements each ----
  {
    int row = tid >> 3, seg = tid & 7;
    float m = -1e30f;
    for (int i = 0; i < 32; i++) m = fmaxf(m, sim[row][seg * 32 + i]);
    for (int o = 4; o > 0; o >>= 1) m = fmaxf(m, __shfl_xor(m, o, 8));
    float ssum = 0.f;
    for (int i = 0; i < 32; i++) {
      float e = __expf(sim[row][seg * 32 + i] - m);
      ssum += e;
      sim[row][seg * 32 + i] = e;
    }
    for (int o = 4; o > 0; o >>= 1) ssum += __shfl_xor(ssum, o, 8);
    float inv = 1.f / ssum;
    for (int i = 0; i < 32; i++)
      attnbuf[row][seg * 32 + i] = (bf16)(sim[row][seg * 32 + i] * inv);
  }
  __syncthreads();

  // ---- out = attn @ v ----
  bf16* ob = aout + ((size_t)b * HW_ + p0) * C_;
  for (int nc = 0; nc < 2; nc++) {
    v8f d[4] = {{}, {}, {}, {}};
    for (int kc = 0; kc < S_; kc += 64) {
      // stage V^T tile [256 c][64 s] (transpose through LDS scatter)
#pragma unroll
      for (int j = 0; j < 8; j++) {
        int fid = j * 256 + tid;            // 2048 float4 reads
        int sl = fid >> 5, c8 = fid & 31;   // s-local, 8-wide c chunk
        float4 pack = *(const float4*)&kvb[(size_t)(kc + sl) * (2 * C_) + C_ +
                                           nc * 256 + c8 * 8];
        const bf16* vals = (const bf16*)&pack;
#pragma unroll
        for (int u = 0; u < 8; u++) Vt[c8 * 8 + u][sl] = vals[u];
      }
      __syncthreads();
#pragma unroll
      for (int kk = 0; kk < 64; kk += 32) {
        v16bf a = frag_a(&attnbuf[wr * 16][kc + kk], 256);
#pragma unroll
        for (int t = 0; t < 4; t++) {
          v16bf bb = frag_b(&Vt[wc * 64 + t * 16][kk], 64);
          d[t] = wmma_bf16(a, bb, d[t]);
        }
      }
      __syncthreads();
    }
    int n = lane & 15, mo = (lane >> 4) * 8;
#pragma unroll
    for (int t = 0; t < 4; t++)
#pragma unroll
      for (int r = 0; r < 8; r++)
        ob[(size_t)(wr * 16 + mo + r) * C_ + nc * 256 + wc * 64 + t * 16 + n] =
            (bf16)d[t][r];
  }
}

// ---- out-projection + bias + residual, writes f32 NCHW --------------------
__global__ __launch_bounds__(256) void outproj_kernel(
    const bf16* __restrict__ A,   // aout [b][4096][512]
    const bf16* __restrict__ W,   // wo [512][512] bf16
    const float* __restrict__ bo, const float* __restrict__ x,
    float* __restrict__ out) {
  __shared__ __align__(16) bf16 As[2][64][64];
  __shared__ __align__(16) bf16 Ws[2][64][64];
  int b = blockIdx.z;
  const bf16* Ab = A + (size_t)b * HW_ * C_;
  int row0 = blockIdx.x * 64, col0 = blockIdx.y * 64;
  int tid = threadIdx.x, lane = tid & 31, wid = tid >> 5;
  int wm = wid & 3, wn = wid >> 2;
  int r = tid >> 2, seg = tid & 3;
  const bf16* arow = Ab + (size_t)(row0 + r) * C_ + seg * 16;
  const bf16* wrow = W + (size_t)(col0 + r) * C_ + seg * 16;

  async_b128(arow,     &As[0][r][seg * 16]);
  async_b128(arow + 8, &As[0][r][seg * 16 + 8]);
  async_b128(wrow,     &Ws[0][r][seg * 16]);
  async_b128(wrow + 8, &Ws[0][r][seg * 16 + 8]);
  v8f c0 = {}, c1 = {};
  const int nch = C_ >> 6;
  wait_async();
  __syncthreads();
  for (int i = 0; i < nch; ++i) {
    const int cur = i & 1;
    if (i + 1 < nch) {
      const bf16* a2 = arow + (i + 1) * 64;
      const bf16* w2 = wrow + (i + 1) * 64;
      async_b128(a2,     &As[cur ^ 1][r][seg * 16]);
      async_b128(a2 + 8, &As[cur ^ 1][r][seg * 16 + 8]);
      async_b128(w2,     &Ws[cur ^ 1][r][seg * 16]);
      async_b128(w2 + 8, &Ws[cur ^ 1][r][seg * 16 + 8]);
    }
    if (i + 2 < nch) __builtin_prefetch(arow + (i + 2) * 64, 0, 1);
#pragma unroll
    for (int kk = 0; kk < 64; kk += 32) {
      v16bf a  = frag_a(&As[cur][wm * 16][kk], 64);
      v16bf b0 = frag_b(&Ws[cur][wn * 32][kk], 64);
      v16bf b1 = frag_b(&Ws[cur][wn * 32 + 16][kk], 64);
      c0 = wmma_bf16(a, b0, c0);
      c1 = wmma_bf16(a, b1, c1);
    }
    wait_async();
    __syncthreads();
  }
  int n = lane & 15, mo = (lane >> 4) * 8;
  int cA = col0 + wn * 32 + n, cB = cA + 16;
  float biasA = bo[cA], biasB = bo[cB];
#pragma unroll
  for (int rr = 0; rr < 8; rr++) {
    size_t p = (size_t)(row0 + wm * 16 + mo + rr);
    size_t iA = ((size_t)b * C_ + cA) * HW_ + p;
    size_t iB = ((size_t)b * C_ + cB) * HW_ + p;
    out[iA] = c0[rr] + biasA + x[iA];
    out[iB] = c1[rr] + biasB + x[iB];
  }
}

// ---------------------------------------------------------------------------
extern "C" void kernel_launch(void* const* d_in, const int* in_sizes, int n_in,
                              void* d_out, int out_size, void* d_ws,
                              size_t ws_size, hipStream_t stream) {
  const float* x    = (const float*)d_in[0];
  const float* ctx  = (const float*)d_in[1];
  const float* gn_w = (const float*)d_in[2];
  const float* gn_b = (const float*)d_in[3];
  const float* ln_w = (const float*)d_in[4];
  const float* ln_b = (const float*)d_in[5];
  const float* wq   = (const float*)d_in[6];
  const float* bq   = (const float*)d_in[7];
  const float* wkv  = (const float*)d_in[8];
  const float* bkv  = (const float*)d_in[9];
  const float* wo   = (const float*)d_in[10];
  const float* bo   = (const float*)d_in[11];
  float* out = (float*)d_out;

  char* ws = (char*)d_ws;
  size_t off = 0;
  auto take = [&](size_t bytes) -> char* {
    char* p = ws + off;
    off += (bytes + 255) & ~(size_t)255;
    return p;
  };
  float* mean  = (float*)take(B_ * GRP_ * sizeof(float));
  float* rstd  = (float*)take(B_ * GRP_ * sizeof(float));
  bf16* wq_b   = (bf16*)take((size_t)C_ * C_ * 2);
  bf16* wkv_b  = (bf16*)take((size_t)2 * C_ * CTX_ * 2);
  bf16* wo_b   = (bf16*)take((size_t)C_ * C_ * 2);
  bf16* cn     = (bf16*)take((size_t)B_ * S_ * CTX_ * 2);
  bf16* kvb    = (bf16*)take((size_t)B_ * S_ * 2 * C_ * 2);
  bf16* xnT    = (bf16*)take((size_t)B_ * HW_ * C_ * 2);
  bf16* qb     = (bf16*)take((size_t)B_ * HW_ * C_ * 2);
  bf16* aout   = xnT;  // xnT is dead after the Q projection; reuse it.

  cvt_bf16_kernel<<<(C_ * C_ + 255) / 256, 256, 0, stream>>>(wq, wq_b, C_ * C_);
  cvt_bf16_kernel<<<(2 * C_ * CTX_ + 255) / 256, 256, 0, stream>>>(
      wkv, wkv_b, 2 * C_ * CTX_);
  cvt_bf16_kernel<<<(C_ * C_ + 255) / 256, 256, 0, stream>>>(wo, wo_b, C_ * C_);

  gn_stats_kernel<<<B_ * GRP_, 256, 0, stream>>>(x, mean, rstd);
  gn_pack_kernel<<<dim3(HW_ / 64, C_ / 64, B_), 256, 0, stream>>>(
      x, gn_w, gn_b, mean, rstd, xnT);
  ln_pack_kernel<<<B_ * S_, 256, 0, stream>>>(ctx, ln_w, ln_b, cn);

  // kv = cn @ wkv^T + bkv : per-batch M=256, N=1024, K=768
  gemm_bf16_nt_kernel<<<dim3(S_ / 64, 2 * C_ / 64, B_), 256, 0, stream>>>(
      cn, (long)S_ * CTX_, wkv_b, bkv, kvb, 2 * C_, (long)S_ * 2 * C_, CTX_);
  // q = xnT @ wq^T + bq : per-batch M=4096, N=512, K=512
  gemm_bf16_nt_kernel<<<dim3(HW_ / 64, C_ / 64, B_), 256, 0, stream>>>(
      xnT, (long)HW_ * C_, wq_b, bq, qb, C_, (long)HW_ * C_, C_);

  attn_kernel<<<dim3(HW_ / 32, B_), 256, 0, stream>>>(qb, kvb, aout);

  outproj_kernel<<<dim3(HW_ / 64, C_ / 64, B_), 256, 0, stream>>>(
      aout, wo_b, bo, x, out);
}
